// CustomBackbone_33165737460313
// MI455X (gfx1250) — matrix operand order repaired and checked by reading, not compiled, not run on hardware
//
#include <hip/hip_runtime.h>
#include <hip/hip_bf16.h>

typedef __attribute__((ext_vector_type(16))) _Float16 v16h;
typedef __attribute__((ext_vector_type(8)))  _Float16 v8h;
typedef __attribute__((ext_vector_type(4)))  _Float16 v4h;
typedef __attribute__((ext_vector_type(8)))  float    v8f;
typedef __attribute__((ext_vector_type(4)))  int      v4i;

#define BN_SCALE 0.99999950000037497f  /* 1/sqrt(1+1e-5) */

// ---------------------------------------------------------------------------
// Async global->LDS staging plumbing (gfx1250). Falls back to load+ds_store.
// Builtin signature (from compiler diagnostic): (v4i AS1*, v4i AS3*, Ii, Ii).
// ---------------------------------------------------------------------------
typedef __attribute__((address_space(1))) v4i* gv4p;
typedef __attribute__((address_space(3))) v4i* lv4p;

#if __has_builtin(__builtin_amdgcn_global_load_async_to_lds_b128)
#define HAVE_ASYNC_LDS 1
__device__ __forceinline__ void async_cp_b128(const _Float16* g, _Float16* l) {
  __builtin_amdgcn_global_load_async_to_lds_b128(
      (gv4p)(size_t)g, (lv4p)(unsigned)(size_t)l, 0, 0);
}
__device__ __forceinline__ void async_wait0() {
#if __has_builtin(__builtin_amdgcn_s_wait_asynccnt)
  __builtin_amdgcn_s_wait_asynccnt(0);
#else
  asm volatile("s_wait_asynccnt 0x0" ::: "memory");
#endif
}
#else
#define HAVE_ASYNC_LDS 0
__device__ __forceinline__ void async_cp_b128(const _Float16* g, _Float16* l) {
  *(v8h*)l = *(const v8h*)g;   // sync fallback: global load + ds store
}
__device__ __forceinline__ void async_wait0() {}
#endif

// ---------------------------------------------------------------------------
// Mask kernels
// ---------------------------------------------------------------------------
__global__ void mask_init_kernel(const float* __restrict__ occ,
                                 float* __restrict__ m0, int n) {
  int i = blockIdx.x * blockDim.x + threadIdx.x;
  if (i < n) m0[i] = (occ[i] < 0.1f) ? 1.0f : 0.0f;
}

__global__ void mask_pool_kernel(const float* __restrict__ in,
                                 float* __restrict__ out, int Do) {
  int i = blockIdx.x * blockDim.x + threadIdx.x;
  int n = 2 * Do * Do * Do;
  if (i >= n) return;
  int b = i / (Do * Do * Do);
  int r = i % (Do * Do * Do);
  int z = r / (Do * Do); r %= Do * Do;
  int y = r / Do;
  int x = r % Do;
  int Di = 2 * Do;
  float m = 0.0f;
  for (int dz = 0; dz < 2; ++dz)
    for (int dy = 0; dy < 2; ++dy)
      for (int dx = 0; dx < 2; ++dx) {
        size_t idx = (((size_t)b * Di + (2 * z + dz)) * Di + (2 * y + dy)) * Di + (2 * x + dx);
        m = fmaxf(m, in[idx]);
      }
  out[i] = m;
}

// ---------------------------------------------------------------------------
// f32 -> f16 weight conversion
// ---------------------------------------------------------------------------
__global__ void cvt_f32_to_f16_kernel(const float* __restrict__ in,
                                      _Float16* __restrict__ out, int n) {
  int i = blockIdx.x * blockDim.x + threadIdx.x;
  if (i < n) out[i] = (_Float16)in[i];
}

// ---------------------------------------------------------------------------
// Block-1 prep: xm4[v] = f16(x[v,0..2] * mask0[v]) channel-padded to 4
// ---------------------------------------------------------------------------
__global__ void xmask_pack_kernel(const float* __restrict__ x,
                                  const float* __restrict__ m0,
                                  _Float16* __restrict__ xm4, int nvox) {
  int i = blockIdx.x * blockDim.x + threadIdx.x;
  if (i >= nvox) return;
  float mk = m0[i];
  const float* px = x + (size_t)i * 3;
  _Float16* o = xm4 + (size_t)i * 4;
  o[0] = (_Float16)(px[0] * mk);
  o[1] = (_Float16)(px[1] * mk);
  o[2] = (_Float16)(px[2] * mk);
  o[3] = (_Float16)0.0f;
}

__global__ void w1_pack_kernel(const float* __restrict__ W1,
                               _Float16* __restrict__ w1p) {
  int i = blockIdx.x * blockDim.x + threadIdx.x;
  if (i >= 128 * 64) return;
  int k = i / 64, co = i % 64;
  int tap = k >> 2, ci = k & 3;
  w1p[i] = (tap < 27 && ci < 3) ? (_Float16)W1[((size_t)tap * 3 + ci) * 64 + co]
                                : (_Float16)0.0f;
}

// ---------------------------------------------------------------------------
// Block 1 implicit GEMM: M = 2*64^3, N = 64, K = 128 (27 taps x 4ch, padded).
// ---------------------------------------------------------------------------
__global__ void conv1_wmma_kernel(const _Float16* __restrict__ xm4,
                                  const _Float16* __restrict__ w1p,
                                  const float* __restrict__ gamma,
                                  const float* __restrict__ beta,
                                  const float* __restrict__ m0,
                                  float* __restrict__ outF32,
                                  _Float16* __restrict__ outF16) {
  constexpr int D = 64, COUT = 64;
  constexpr int M = 2 * D * D * D;
  constexpr int MT = M / 16;
  int wave = blockIdx.x * (blockDim.x >> 5) + (threadIdx.x >> 5);
  int lane = threadIdx.x & 31;
  if (wave >= MT) return;
  int mtile = wave;

  int mA = mtile * 16 + (lane & 15);
  int b  = mA / (D * D * D);
  int r  = mA % (D * D * D);
  int z  = r / (D * D); r %= D * D;
  int y  = r / D;
  int xv = r % D;
  const int koffA = (lane >= 16) ? 8 : 0;
  const int kB    = (lane & 15) + ((lane >= 16) ? 16 : 0);

  v8f acc0 = {}, acc1 = {}, acc2 = {}, acc3 = {};
#pragma unroll
  for (int c0 = 0; c0 < 128; c0 += 32) {
    union { v16h v; v4h g[4]; } ua;
#pragma unroll
    for (int gi = 0; gi < 4; ++gi) {
      int k0  = c0 + koffA + ((gi < 2) ? gi * 4 : 16 + (gi - 2) * 4);
      int tap = k0 >> 2;
      v4h val = {};
      if (tap < 27) {
        int kd = tap / 9, kh = (tap % 9) / 3, kw = tap % 3;
        int zi = z - 1 + kd, yi = y - 1 + kh, xi = xv - 1 + kw;
        if (zi >= 0 && zi < D && yi >= 0 && yi < D && xi >= 0 && xi < D) {
          size_t vi = (((size_t)b * D + zi) * D + yi) * D + xi;
          val = *(const v4h*)(xm4 + vi * 4);
        }
      }
      ua.g[gi] = val;
    }
    const _Float16* pbk = w1p + (size_t)c0 * COUT + (lane & 15);
    union { v16h v; _Float16 e[16]; } ub0, ub1, ub2, ub3;
#pragma unroll
    for (int s2 = 0; s2 < 16; ++s2) {
      int kk = koffA + ((s2 < 8) ? s2 : 8 + s2);
      ub0.e[s2] = pbk[kk * COUT +  0];
      ub1.e[s2] = pbk[kk * COUT + 16];
      ub2.e[s2] = pbk[kk * COUT + 32];
      ub3.e[s2] = pbk[kk * COUT + 48];
    }

    acc0 = __builtin_amdgcn_wmma_f32_16x16x32_f16(false, ua.v, false, ub0.v, (short)0, acc0, false, false);
    acc1 = __builtin_amdgcn_wmma_f32_16x16x32_f16(false, ua.v, false, ub1.v, (short)0, acc1, false, false);
    acc2 = __builtin_amdgcn_wmma_f32_16x16x32_f16(false, ua.v, false, ub2.v, (short)0, acc2, false, false);
    acc3 = __builtin_amdgcn_wmma_f32_16x16x32_f16(false, ua.v, false, ub3.v, (short)0, acc3, false, false);
  }

  int n = lane & 15;
  float gj[4], bj[4];
#pragma unroll
  for (int j = 0; j < 4; ++j) { gj[j] = gamma[j * 16 + n]; bj[j] = beta[j * 16 + n]; }
  int mbase = mtile * 16 + ((lane >= 16) ? 8 : 0);
#pragma unroll
  for (int rr = 0; rr < 8; ++rr) {
    int m = mbase + rr;
    float mk = m0[m];
#pragma unroll
    for (int j = 0; j < 4; ++j) {
      float a = (j == 0) ? acc0[rr] : (j == 1) ? acc1[rr] : (j == 2) ? acc2[rr] : acc3[rr];
      float v = gj[j] * (a * BN_SCALE) + bj[j];
      v = fmaxf(v, 0.0f) * mk;
      size_t oi = (size_t)m * COUT + j * 16 + n;
      outF32[oi] = v;
      outF16[oi] = (_Float16)v;
    }
  }
}

// ---------------------------------------------------------------------------
// Blocks 2-5: implicit-GEMM stride-2 conv. Workgroup = 8 waves sharing one
// 64-cout n-slice; weights double-buffer-staged in LDS via async copies.
// ---------------------------------------------------------------------------
template <int CIN, int COUT, int DI, int DO>
__global__ void conv_block_wmma_kernel(const _Float16* __restrict__ inF16,  // [2,DI^3,CIN]
                                       const _Float16* __restrict__ wF16,   // [27,CIN,COUT]
                                       const float* __restrict__ gamma,
                                       const float* __restrict__ beta,
                                       const float* __restrict__ maskOut,   // [2,DO^3]
                                       float* __restrict__ outF32,
                                       _Float16* __restrict__ outF16) {
  constexpr int M   = 2 * DO * DO * DO;
  constexpr int MT  = M / 16;
  constexpr int CH  = CIN / 32;     // k-chunks per tap
  constexpr int S   = 27 * CH;      // staged tiles

  __shared__ __align__(16) _Float16 sB[2][32 * 64];   // 2 x 4KB

  int tid    = threadIdx.x;
  int waveid = tid >> 5;
  int lane   = tid & 31;
  int ntile4 = blockIdx.x / (MT / 8);
  int mtile  = (blockIdx.x % (MT / 8)) * 8 + waveid;

  // This thread's slot in the 32x64 staged weight tile
  int srow = tid >> 3;
  int scol = (tid & 7) * 8;
  const _Float16* wbase = wF16 + (size_t)ntile4 * 64 + scol;
  _Float16* lslot0 = &sB[0][srow * 64 + scol];
  _Float16* lslot1 = &sB[1][srow * 64 + scol];

  // Output-voxel decode for this lane's A row
  int mA = mtile * 16 + (lane & 15);
  int b  = mA / (DO * DO * DO);
  int r  = mA % (DO * DO * DO);
  int zo = r / (DO * DO); r %= DO * DO;
  int yo = r / DO;
  int xo = r % DO;
  const int koffA = (lane >= 16) ? 8 : 0;
  const int kB    = (lane & 15) + ((lane >= 16) ? 16 : 0);

  auto stage = [&](int s, int bi) {
    int tap = s / CH;
    int c0  = (s & (CH - 1)) * 32;
    const _Float16* g = wbase + ((size_t)tap * CIN + c0 + srow) * COUT;
    async_cp_b128(g, bi ? lslot1 : lslot0);
  };

  v8f acc0 = {}, acc1 = {}, acc2 = {}, acc3 = {};
  stage(0, 0);
  async_wait0();
  __syncthreads();

#pragma unroll 1
  for (int s = 0; s < S; ++s) {
    int p = s & 1;
    if (s + 1 < S) stage(s + 1, p ^ 1);

    int tap = s / CH;            // wave-uniform
    int c0  = (s & (CH - 1)) * 32;
    int kd = tap / 9, kh = (tap % 9) / 3, kw = tap % 3;
    int zi = zo * 2 - 1 + kd, yi = yo * 2 - 1 + kh, xi = xo * 2 - 1 + kw;
    bool valid = (zi >= 0) && (zi < DI) && (yi >= 0) && (yi < DI) &&
                 (xi >= 0) && (xi < DI);

    // A fragment: two aligned 16B global loads, predicated per lane
    v16h afrag = {};
    if (valid) {
      const _Float16* pa =
          inF16 + ((((size_t)b * DI + zi) * DI + yi) * DI + xi) * CIN + c0 + koffA;
      union { v16h v; v8h h[2]; } ua;
      ua.h[0] = *(const v8h*)(pa);
      ua.h[1] = *(const v8h*)(pa + 16);
      afrag = ua.v;
    }

    // B fragments from LDS: lane's cout column n, elements over k in table order
    const _Float16* lb = &sB[p][lane & 15];
    union { v16h v; _Float16 e[16]; } ub0, ub1, ub2, ub3;
#pragma unroll
    for (int s2 = 0; s2 < 16; ++s2) {
      int kk = koffA + ((s2 < 8) ? s2 : 8 + s2);
      ub0.e[s2] = lb[kk * 64 +  0];
      ub1.e[s2] = lb[kk * 64 + 16];
      ub2.e[s2] = lb[kk * 64 + 32];
      ub3.e[s2] = lb[kk * 64 + 48];
    }

    acc0 = __builtin_amdgcn_wmma_f32_16x16x32_f16(false, afrag, false, ub0.v, (short)0, acc0, false, false);
    acc1 = __builtin_amdgcn_wmma_f32_16x16x32_f16(false, afrag, false, ub1.v, (short)0, acc1, false, false);
    acc2 = __builtin_amdgcn_wmma_f32_16x16x32_f16(false, afrag, false, ub2.v, (short)0, acc2, false, false);
    acc3 = __builtin_amdgcn_wmma_f32_16x16x32_f16(false, afrag, false, ub3.v, (short)0, acc3, false, false);

    async_wait0();      // next tile staged
    __syncthreads();    // all waves done reading buffer p, staging visible
  }

  // ---- Epilogue: BN(eval) + ReLU + output mask
  int n = lane & 15;
  float gj[4], bj[4];
#pragma unroll
  for (int j = 0; j < 4; ++j) {
    int co = (ntile4 * 4 + j) * 16 + n;
    gj[j] = gamma[co];
    bj[j] = beta[co];
  }
  int mbase = mtile * 16 + ((lane >= 16) ? 8 : 0);
#pragma unroll
  for (int rr = 0; rr < 8; ++rr) {
    int m = mbase + rr;
    float mk = maskOut[m];
#pragma unroll
    for (int j = 0; j < 4; ++j) {
      float a = (j == 0) ? acc0[rr] : (j == 1) ? acc1[rr] : (j == 2) ? acc2[rr] : acc3[rr];
      float v = gj[j] * (a * BN_SCALE) + bj[j];
      v = fmaxf(v, 0.0f) * mk;
      size_t oi = (size_t)m * COUT + (ntile4 * 4 + j) * 16 + n;
      outF32[oi] = v;
      if (outF16) outF16[oi] = (_Float16)v;
    }
  }
}

// ---------------------------------------------------------------------------
// Host side
// ---------------------------------------------------------------------------
static inline size_t ws_alloc(size_t& off, size_t bytes) {
  size_t o = off;
  off = (off + bytes + 255) & ~(size_t)255;
  return o;
}

extern "C" void kernel_launch(void* const* d_in, const int* in_sizes, int n_in,
                              void* d_out, int out_size, void* d_ws, size_t ws_size,
                              hipStream_t stream) {
  (void)in_sizes; (void)n_in; (void)out_size; (void)ws_size;

  const float* x   = (const float*)d_in[0];
  const float* occ = (const float*)d_in[1];
  const float* W1  = (const float*)d_in[2];
  const float* g1  = (const float*)d_in[3];
  const float* b1  = (const float*)d_in[4];
  const float* W2  = (const float*)d_in[5];
  const float* g2  = (const float*)d_in[6];
  const float* b2  = (const float*)d_in[7];
  const float* W3  = (const float*)d_in[8];
  const float* g3  = (const float*)d_in[9];
  const float* b3  = (const float*)d_in[10];
  const float* W4  = (const float*)d_in[11];
  const float* g4  = (const float*)d_in[12];
  const float* b4  = (const float*)d_in[13];
  const float* W5  = (const float*)d_in[14];
  const float* g5  = (const float*)d_in[15];
  const float* b5  = (const float*)d_in[16];

  float* out = (float*)d_out;
  const size_t o1 = 0;
  const size_t o2 = o1 + (size_t)2 * 64 * 64 * 64 * 64;
  const size_t o3 = o2 + (size_t)2 * 32 * 32 * 32 * 128;
  const size_t o4 = o3 + (size_t)2 * 16 * 16 * 16 * 256;
  const size_t o5 = o4 + (size_t)2 * 8 * 8 * 8 * 512;

  char* ws = (char*)d_ws;
  size_t off = 0;
  const size_t nV0 = (size_t)2 * 64 * 64 * 64;
  float* m0 = (float*)(ws + ws_alloc(off, nV0 * 4));
  float* m2 = (float*)(ws + ws_alloc(off, (size_t)2 * 32 * 32 * 32 * 4));
  float* m3 = (float*)(ws + ws_alloc(off, (size_t)2 * 16 * 16 * 16 * 4));
  float* m4 = (float*)(ws + ws_alloc(off, (size_t)2 * 8 * 8 * 8 * 4));
  float* m5 = (float*)(ws + ws_alloc(off, (size_t)2 * 4 * 4 * 4 * 4));
  _Float16* x1h = (_Float16*)(ws + ws_alloc(off, (size_t)2 * 64 * 64 * 64 * 64 * 2));
  _Float16* x2h = (_Float16*)(ws + ws_alloc(off, (size_t)2 * 32 * 32 * 32 * 128 * 2));
  _Float16* x3h = (_Float16*)(ws + ws_alloc(off, (size_t)2 * 16 * 16 * 16 * 256 * 2));
  _Float16* x4h = (_Float16*)(ws + ws_alloc(off, (size_t)2 * 8 * 8 * 8 * 512 * 2));
  const int nW2 = 27 * 64 * 128, nW3 = 27 * 128 * 256,
            nW4 = 27 * 256 * 512, nW5 = 27 * 512 * 1024;
  _Float16* W2h = (_Float16*)(ws + ws_alloc(off, (size_t)nW2 * 2));
  _Float16* W3h = (_Float16*)(ws + ws_alloc(off, (size_t)nW3 * 2));
  _Float16* W4h = (_Float16*)(ws + ws_alloc(off, (size_t)nW4 * 2));
  _Float16* W5h = (_Float16*)(ws + ws_alloc(off, (size_t)nW5 * 2));
  _Float16* xm4 = (_Float16*)(ws + ws_alloc(off, nV0 * 4 * 2));
  _Float16* w1p = (_Float16*)(ws + ws_alloc(off, (size_t)128 * 64 * 2));

  // 1) masks + block-1 packed input/weights
  {
    int n = (int)nV0;
    mask_init_kernel<<<(n + 255) / 256, 256, 0, stream>>>(occ, m0, n);
    mask_pool_kernel<<<(2 * 32 * 32 * 32 + 255) / 256, 256, 0, stream>>>(m0, m2, 32);
    mask_pool_kernel<<<(2 * 16 * 16 * 16 + 255) / 256, 256, 0, stream>>>(m2, m3, 16);
    mask_pool_kernel<<<(2 * 8 * 8 * 8 + 255) / 256, 256, 0, stream>>>(m3, m4, 8);
    mask_pool_kernel<<<(2 * 4 * 4 * 4 + 255) / 256, 256, 0, stream>>>(m4, m5, 4);
    xmask_pack_kernel<<<(n + 255) / 256, 256, 0, stream>>>(x, m0, xm4, n);
    w1_pack_kernel<<<(128 * 64 + 255) / 256, 256, 0, stream>>>(W1, w1p);
  }

  // 2) weight conversion f32 -> f16
  cvt_f32_to_f16_kernel<<<(nW2 + 255) / 256, 256, 0, stream>>>(W2, W2h, nW2);
  cvt_f32_to_f16_kernel<<<(nW3 + 255) / 256, 256, 0, stream>>>(W3, W3h, nW3);
  cvt_f32_to_f16_kernel<<<(nW4 + 255) / 256, 256, 0, stream>>>(W4, W4h, nW4);
  cvt_f32_to_f16_kernel<<<(nW5 + 255) / 256, 256, 0, stream>>>(W5, W5h, nW5);

  // 3) block 1 (WMMA over padded K=128)
  {
    int waves = (int)(nV0 / 16);
    conv1_wmma_kernel<<<(waves + 7) / 8, 256, 0, stream>>>(xm4, w1p, g1, b1, m0,
                                                           out + o1, x1h);
  }

  // 4) blocks 2-5: 8 waves per block sharing one LDS-staged weight slice
  {
    int blocks = (2 * 32 * 32 * 32 / 16 / 8) * (128 / 64);
    conv_block_wmma_kernel<64, 128, 64, 32>
        <<<blocks, 256, 0, stream>>>(x1h, W2h, g2, b2, m2, out + o2, x2h);
  }
  {
    int blocks = (2 * 16 * 16 * 16 / 16 / 8) * (256 / 64);
    conv_block_wmma_kernel<128, 256, 32, 16>
        <<<blocks, 256, 0, stream>>>(x2h, W3h, g3, b3, m3, out + o3, x3h);
  }
  {
    int blocks = (2 * 8 * 8 * 8 / 16 / 8) * (512 / 64);
    conv_block_wmma_kernel<256, 512, 16, 8>
        <<<blocks, 256, 0, stream>>>(x3h, W4h, g4, b4, m4, out + o4, x4h);
  }
  {
    int blocks = (2 * 4 * 4 * 4 / 16 / 8) * (1024 / 64);
    conv_block_wmma_kernel<512, 1024, 8, 4>
        <<<blocks, 256, 0, stream>>>(x4h, W5h, g5, b5, m5, out + o5,
                                     (_Float16*)nullptr);
  }
}
